// DeltaNetLayer_22488448762134
// MI455X (gfx1250) — compile-verified, matrix operand order checked by reading, not run on hardware
//
#include <hip/hip_runtime.h>
#include <hip/hip_bf16.h>

#define HH 16
#define DD 64
#define EE 1024
#define WIN 64
#define CHK 64

typedef __attribute__((ext_vector_type(16))) __bf16 v16bf;
typedef __attribute__((ext_vector_type(8)))  __bf16 v8bf;
typedef __attribute__((ext_vector_type(8)))  float  v8f;

__device__ __forceinline__ unsigned short f32_to_bf16_rne(float f) {
    unsigned int u = __float_as_uint(f);
    unsigned int r = u + 0x7FFFu + ((u >> 16) & 1u);
    return (unsigned short)(r >> 16);
}

// Load one 16x32 bf16 WMMA fragment (per-lane K pattern per ISA: lane<16 holds
// K={0..7,16..23}, lane>=16 holds K={8..15,24..31}; caller pre-offsets by hi*8).
__device__ __forceinline__ v16bf ldfrag(const unsigned short* p) {
    v8bf lo = *(const v8bf*)p;
    v8bf hi = *(const v8bf*)(p + 16);
    return __builtin_shufflevector(lo, hi, 0,1,2,3,4,5,6,7,8,9,10,11,12,13,14,15);
}

// ---------------- conversion kernels ----------------
__global__ void cvt_bf16_kernel(const float* __restrict__ in,
                                unsigned short* __restrict__ out, int n) {
    int i = blockIdx.x * blockDim.x + threadIdx.x;
    if (i < n) out[i] = f32_to_bf16_rne(in[i]);
}

// in: [K][N] row-major f32  ->  out: [N][K] bf16 (transposed)
__global__ void cvt_bf16_transpose_kernel(const float* __restrict__ in,
                                          unsigned short* __restrict__ out,
                                          int K, int N) {
    int i = blockIdx.x * blockDim.x + threadIdx.x;
    if (i < K * N) {
        int k = i / N, n = i % N;
        out[(size_t)n * K + k] = f32_to_bf16_rne(in[i]);
    }
}

// ---------------- WMMA GEMM: C[MxN](f32) = A_bf16[MxK] @ Bt_bf16[NxK]^T ----------------
// Register-blocked: each wave computes a 32x64 tile (2 A-frags x 4 B-frags,
// 8 v8f accumulators) -> 6 fragment loads feed 8 WMMAs per K-step.
// Block = 8 waves (2 row x 4 col) -> 64x256 tile.
// mode 0: C = acc
// mode 1: C = X + (acc + bias[n]) * gamma[n]     (final projection epilogue)
__global__ __launch_bounds__(256)
void wmma_gemm_bf16(const unsigned short* __restrict__ A,
                    const unsigned short* __restrict__ Bt,
                    float* __restrict__ C,
                    const float* __restrict__ X,
                    const float* __restrict__ bias,
                    const float* __restrict__ gamma,
                    int M, int N, int K, int mode) {
    const int lane  = threadIdx.x & 31;
    const int wave  = threadIdx.x >> 5;        // 0..7
    const int waveR = wave >> 2;               // 0..1
    const int waveC = wave & 3;                // 0..3
    const int rowBlk = blockIdx.x * 64  + waveR * 32;
    const int colBlk = blockIdx.y * 256 + waveC * 64;
    const int hi  = lane >> 4;                 // 0|1
    const int l15 = lane & 15;

    const unsigned short* pa0 = A + (size_t)(rowBlk + l15) * K + hi * 8;
    const unsigned short* pa1 = pa0 + (size_t)16 * K;
    const unsigned short* pb0 = Bt + (size_t)(colBlk +  0 + l15) * K + hi * 8;
    const unsigned short* pb1 = Bt + (size_t)(colBlk + 16 + l15) * K + hi * 8;
    const unsigned short* pb2 = Bt + (size_t)(colBlk + 32 + l15) * K + hi * 8;
    const unsigned short* pb3 = Bt + (size_t)(colBlk + 48 + l15) * K + hi * 8;

    v8f acc[2][4];
    #pragma unroll
    for (int r = 0; r < 2; ++r)
        #pragma unroll
        for (int c = 0; c < 4; ++c) acc[r][c] = (v8f){};

    for (int k0 = 0; k0 < K; k0 += 32) {
        v16bf a0 = ldfrag(pa0 + k0);
        v16bf a1 = ldfrag(pa1 + k0);
        v16bf b0 = ldfrag(pb0 + k0);
        v16bf b1 = ldfrag(pb1 + k0);
        v16bf b2 = ldfrag(pb2 + k0);
        v16bf b3 = ldfrag(pb3 + k0);
        acc[0][0] = __builtin_amdgcn_wmma_f32_16x16x32_bf16(false, a0, false, b0, (short)0, acc[0][0], false, false);
        acc[1][0] = __builtin_amdgcn_wmma_f32_16x16x32_bf16(false, a1, false, b0, (short)0, acc[1][0], false, false);
        acc[0][1] = __builtin_amdgcn_wmma_f32_16x16x32_bf16(false, a0, false, b1, (short)0, acc[0][1], false, false);
        acc[1][1] = __builtin_amdgcn_wmma_f32_16x16x32_bf16(false, a1, false, b1, (short)0, acc[1][1], false, false);
        acc[0][2] = __builtin_amdgcn_wmma_f32_16x16x32_bf16(false, a0, false, b2, (short)0, acc[0][2], false, false);
        acc[1][2] = __builtin_amdgcn_wmma_f32_16x16x32_bf16(false, a1, false, b2, (short)0, acc[1][2], false, false);
        acc[0][3] = __builtin_amdgcn_wmma_f32_16x16x32_bf16(false, a0, false, b3, (short)0, acc[0][3], false, false);
        acc[1][3] = __builtin_amdgcn_wmma_f32_16x16x32_bf16(false, a1, false, b3, (short)0, acc[1][3], false, false);
    }
    // C/D layout: VGPR r -> M = r + 8*hi, N = l15
    #pragma unroll
    for (int r2 = 0; r2 < 2; ++r2) {
        #pragma unroll
        for (int c = 0; c < 4; ++c) {
            #pragma unroll
            for (int r = 0; r < 8; ++r) {
                int row = rowBlk + r2 * 16 + r + hi * 8;
                int col = colBlk + c * 16 + l15;
                size_t idx = (size_t)row * N + col;
                float v = acc[r2][c][r];
                if (mode == 1) v = X[idx] + (v + bias[col]) * gamma[col];
                C[idx] = v;
            }
        }
    }
}

// ---------------- RoPE (+scale on q) and gate/fuse sigmoid projections ----------------
__global__ __launch_bounds__(256)
void rope_gates_kernel(const float* __restrict__ x,
                       float* __restrict__ q, float* __restrict__ k,
                       const float* __restrict__ Wg, const float* __restrict__ bg,
                       const float* __restrict__ Wf, const float* __restrict__ bf,
                       float* __restrict__ gate, float* __restrict__ fuse, int T) {
    const int tok = blockIdx.x;           // b*T + t
    const int t   = tok % T;
    const int tid = threadIdx.x;
    float* qrow = q + (size_t)tok * EE;
    float* krow = k + (size_t)tok * EE;
    const float LOG1E4_OVER_HALF = 0.28782313662425572f;  // ln(10000)/32
    for (int p = tid; p < EE / 2; p += 256) {
        int e  = p * 2;
        int pi = (e & 63) >> 1;           // pair index within head, 0..31
        float ang = (float)t * __expf(-(float)pi * LOG1E4_OVER_HALF);
        float c = __cosf(ang), s = __sinf(ang);
        float q1 = qrow[e], q2 = qrow[e + 1];
        qrow[e]     = (q1 * c - q2 * s) * 0.125f;   // * D^-0.5
        qrow[e + 1] = (q2 * c + q1 * s) * 0.125f;
        float k1 = krow[e], k2 = krow[e + 1];
        krow[e]     = k1 * c - k2 * s;
        krow[e + 1] = k2 * c + k1 * s;
    }
    if (tid < 32) {
        int h = tid & 15;
        const float* W  = (tid < 16) ? Wg : Wf;
        const float* bb = (tid < 16) ? bg : bf;
        const float* xr = x + (size_t)tok * EE;
        float acc = bb[h];
        for (int e = 0; e < EE; ++e) acc += xr[e] * W[e * HH + h];
        float sgm = 1.f / (1.f + __expf(-acc));
        if (tid < 16) gate[tok * HH + h] = sgm;
        else          fuse[tok * HH + h] = sgm;
    }
}

// ---------------- chunked-parallel gated delta scan ----------------
// Recurrence: S_t = geff_t*S_{t-1} + m_t*k_t v_t^T ; y_t = m_t * q_t^T S_t.
// With intra-chunk cumulative decay a_i = prod_{r<=i} geff_r:
//   y_i = m_i * ( a_i * q_i @ S_prev  +  sum_{s<=i} (a_i/a_s) m_s (q_i.k_s) v_s )
//   S_new = a_63 * S_prev + sum_s (a_63/a_s) m_s k_s v_s^T
// One block per (b,h), 256 threads (quarter = tid>>6 owns 16 rows, e = tid&63
// owns one column). Sequential depth drops 2048 -> 32.
__global__ __launch_bounds__(256)
void delta_chunk_kernel(const float* __restrict__ q, const float* __restrict__ k,
                        const float* __restrict__ v, const float* __restrict__ gate,
                        const float* __restrict__ mask, float* __restrict__ yd, int T) {
    const int b = blockIdx.x >> 4, h = blockIdx.x & 15;
    const int tid = threadIdx.x;
    const int quarter = tid >> 6;         // 0..3
    const int e = tid & 63;               // column / s-index

    __shared__ _Float16 qs[CHK][DD];          // 8 KB (broadcast reads)
    __shared__ float    ks[CHK][DD + 1];      // 16.25 KB (+1 pad: conflict-free ks[e][d])
    __shared__ _Float16 vs[CHK][DD];          // 8 KB
    __shared__ float    S[DD][DD];            // 16 KB state (S[d][e]: lane-consecutive)
    __shared__ _Float16 p[CHK][CHK];          // 8 KB decayed scores
    __shared__ float    a[CHK], mm[CHK], uu[CHK];

    #pragma unroll
    for (int dd2 = 0; dd2 < 16; ++dd2) S[quarter * 16 + dd2][e] = 0.f;
    __syncthreads();

    const int nc = T / CHK;
    for (int c = 0; c < nc; ++c) {
        const int t0 = c * CHK;
        // ---- load chunk ----
        #pragma unroll 4
        for (int ii = 0; ii < 16; ++ii) {
            int i = quarter * 16 + ii;
            size_t base = ((size_t)(b * T + t0 + i) * HH + h) * DD;
            qs[i][e] = (_Float16)q[base + e];
            ks[i][e] = k[base + e];
            vs[i][e] = (_Float16)v[base + e];
        }
        if (tid < CHK) {
            float m = mask[b * T + t0 + tid];
            float g = gate[(b * T + t0 + tid) * HH + h];
            mm[tid] = m;
            a[tid]  = g * m + (1.f - m);       // geff; prefixed below
        }
        __syncthreads();
        if (tid == 0) {
            float run = 1.f;
            for (int i = 0; i < CHK; ++i) { run *= a[i]; a[i] = run; }
        }
        __syncthreads();
        if (tid < CHK) uu[tid] = (a[CHK - 1] / a[tid]) * mm[tid];
        // ---- decayed causal scores: p[i][e] ----
        #pragma unroll 2
        for (int ii = 0; ii < 16; ++ii) {
            int i = quarter * 16 + ii;
            float sdot = 0.f;
            if (e <= i) {
                #pragma unroll
                for (int d = 0; d < DD; ++d) sdot += (float)qs[i][d] * ks[e][d];
                sdot *= (a[i] / a[e]) * mm[e];
            }
            p[i][e] = (_Float16)sdot;
        }
        __syncthreads();
        // ---- y = a_i * Q@S_prev + P@V  (reads old S) ----
        #pragma unroll 2
        for (int ii = 0; ii < 16; ++ii) {
            int i = quarter * 16 + ii;
            float y0 = 0.f;
            #pragma unroll
            for (int d = 0; d < DD; ++d) y0 += (float)qs[i][d] * S[d][e];
            y0 *= a[i];
            float pv = 0.f;
            #pragma unroll
            for (int s = 0; s < CHK; ++s) pv += (float)p[i][s] * (float)vs[s][e];
            size_t base = ((size_t)(b * T + t0 + i) * HH + h) * DD;
            yd[base + e] = (y0 + pv) * mm[i];
        }
        __syncthreads();
        // ---- state update: S = a_63*S + (u o K)^T V ----
        const float aC = a[CHK - 1];
        #pragma unroll 2
        for (int dd2 = 0; dd2 < 16; ++dd2) {
            int d = quarter * 16 + dd2;
            float accS = S[d][e] * aC;
            #pragma unroll
            for (int s = 0; s < CHK; ++s) accS += uu[s] * ks[s][d] * (float)vs[s][e];
            S[d][e] = accS;
        }
        __syncthreads();
    }
}

// ---------------- local windowed attention: one block per (b,h,chunk) ----------------
__global__ __launch_bounds__(64)
void local_attn_kernel(const float* __restrict__ q, const float* __restrict__ k,
                       const float* __restrict__ v, const float* __restrict__ mask,
                       float* __restrict__ yl, int T) {
    const int nc = T / WIN;
    int blk = blockIdx.x;
    int c = blk % nc; int h = (blk / nc) & 15; int b = blk / (nc * HH);
    const int i = threadIdx.x;            // query row within chunk
    __shared__ float     ks_[2 * WIN][DD];
    __shared__ _Float16  vs_[2 * WIN][DD];
    __shared__ float     ms_[2 * WIN];
    for (int j = i; j < 2 * WIN; j += 64) {
        int tj = (j < WIN) ? (c - 1) * WIN + j : c * WIN + (j - WIN);
        bool valid = (j >= WIN) || (c > 0);
        size_t base = valid ? ((size_t)(b * T + tj) * HH + h) * DD : 0;
        ms_[j] = valid ? mask[b * T + tj] : 0.f;
        for (int d = 0; d < DD; ++d) {
            ks_[j][d] = valid ? k[base + d] : 0.f;
            vs_[j][d] = (_Float16)(valid ? v[base + d] : 0.f);
        }
    }
    __syncthreads();
    int tq = c * WIN + i;
    size_t qbase = ((size_t)(b * T + tq) * HH + h) * DD;
    float qr[DD], out[DD];
    #pragma unroll
    for (int d = 0; d < DD; ++d) { qr[d] = q[qbase + d]; out[d] = 0.f; }
    float Mx = -3.0e38f, S = 0.f;
    for (int j = 0; j < 2 * WIN; ++j) {
        bool ok = (j >= i + 1) && (j <= i + WIN) && ((c > 0) || (j >= WIN)) && (ms_[j] > 0.5f);
        float s = 0.f;
        #pragma unroll
        for (int d = 0; d < DD; ++d) s += qr[d] * ks_[j][d];
        s = ok ? s : -1.0e30f;
        float Mn = fmaxf(Mx, s);
        float scale = __expf(Mx - Mn);
        float pw    = __expf(s  - Mn);
        S = S * scale + pw;
        #pragma unroll
        for (int d = 0; d < DD; ++d) out[d] = out[d] * scale + pw * (float)vs_[j][d];
        Mx = Mn;
    }
    float inv = (S > 0.f) ? 1.f / S : 0.f;
    float mi = mask[b * T + tq];
    for (int d = 0; d < DD; ++d) yl[qbase + d] = out[d] * inv * mi;
}

// ---------------- fuse-combine + RMSNorm -> bf16 ----------------
__global__ __launch_bounds__(256)
void combine_norm_kernel(const float* __restrict__ yd, const float* __restrict__ yl,
                         const float* __restrict__ fuse, const float* __restrict__ norm_w,
                         unsigned short* __restrict__ ybf) {
    int tok = blockIdx.x;
    int tid = threadIdx.x;
    __shared__ float red[256];
    float vals[4];
    float ss = 0.f;
    #pragma unroll
    for (int r = 0; r < 4; ++r) {
        int e = tid + r * 256;
        int h = e >> 6;
        float f = fuse[tok * HH + h];
        float y = f * yd[(size_t)tok * EE + e] + (1.f - f) * yl[(size_t)tok * EE + e];
        vals[r] = y;
        ss += y * y;
    }
    red[tid] = ss;
    __syncthreads();
    for (int off = 128; off > 0; off >>= 1) {
        if (tid < off) red[tid] += red[tid + off];
        __syncthreads();
    }
    float rms = rsqrtf(red[0] / (float)EE + 1e-6f);
    #pragma unroll
    for (int r = 0; r < 4; ++r) {
        int e = tid + r * 256;
        ybf[(size_t)tok * EE + e] = f32_to_bf16_rne(vals[r] * rms * norm_w[e]);
    }
}

// ---------------- host launch ----------------
extern "C" void kernel_launch(void* const* d_in, const int* in_sizes, int n_in,
                              void* d_out, int out_size, void* d_ws, size_t ws_size,
                              hipStream_t stream) {
    const int B = 2, T = 2048, BT = B * T;
    const float* x    = (const float*)d_in[0];
    const float* am   = (const float*)d_in[1];
    const float* Wq   = (const float*)d_in[2];
    const float* Wk   = (const float*)d_in[3];
    const float* Wv   = (const float*)d_in[4];
    const float* Wo   = (const float*)d_in[5];
    const float* bo   = (const float*)d_in[6];
    const float* Wg   = (const float*)d_in[7];
    const float* bg   = (const float*)d_in[8];
    const float* Wf   = (const float*)d_in[9];
    const float* bf   = (const float*)d_in[10];
    const float* nw   = (const float*)d_in[11];
    const float* gam  = (const float*)d_in[12];
    float* out = (float*)d_out;

    char* w = (char*)d_ws;
    unsigned short* xbf = (unsigned short*)w;  w += (size_t)BT * EE * 2;
    unsigned short* WqT = (unsigned short*)w;  w += (size_t)EE * EE * 2;
    unsigned short* WkT = (unsigned short*)w;  w += (size_t)EE * EE * 2;
    unsigned short* WvT = (unsigned short*)w;  w += (size_t)EE * EE * 2;
    unsigned short* WoT = (unsigned short*)w;  w += (size_t)EE * EE * 2;
    float* qb   = (float*)w;                   w += (size_t)BT * EE * 4;
    float* kb   = (float*)w;                   w += (size_t)BT * EE * 4;
    float* vb   = (float*)w;                   w += (size_t)BT * EE * 4;
    float* gate = (float*)w;                   w += (size_t)BT * HH * 4;
    float* fuse = (float*)w;                   w += (size_t)BT * HH * 4;
    float* yd   = (float*)w;                   w += (size_t)BT * EE * 4;
    float* yl   = (float*)w;                   w += (size_t)BT * EE * 4;
    unsigned short* ybf = (unsigned short*)w;  w += (size_t)BT * EE * 2;

    // 1. conversions
    {
        int n = BT * EE;
        cvt_bf16_kernel<<<(n + 255) / 256, 256, 0, stream>>>(x, xbf, n);
        int nw2 = EE * EE;
        cvt_bf16_transpose_kernel<<<(nw2 + 255) / 256, 256, 0, stream>>>(Wq, WqT, EE, EE);
        cvt_bf16_transpose_kernel<<<(nw2 + 255) / 256, 256, 0, stream>>>(Wk, WkT, EE, EE);
        cvt_bf16_transpose_kernel<<<(nw2 + 255) / 256, 256, 0, stream>>>(Wv, WvT, EE, EE);
        cvt_bf16_transpose_kernel<<<(nw2 + 255) / 256, 256, 0, stream>>>(Wo, WoT, EE, EE);
    }
    // 2. Q/K/V projections (WMMA bf16 -> f32), 64x256 block tiles
    {
        dim3 grid(BT / 64, EE / 256);
        wmma_gemm_bf16<<<grid, 256, 0, stream>>>(xbf, WqT, qb, nullptr, nullptr, nullptr, BT, EE, EE, 0);
        wmma_gemm_bf16<<<grid, 256, 0, stream>>>(xbf, WkT, kb, nullptr, nullptr, nullptr, BT, EE, EE, 0);
        wmma_gemm_bf16<<<grid, 256, 0, stream>>>(xbf, WvT, vb, nullptr, nullptr, nullptr, BT, EE, EE, 0);
    }
    // 3. RoPE + gates
    rope_gates_kernel<<<BT, 256, 0, stream>>>(x, qb, kb, Wg, bg, Wf, bf, gate, fuse, T);
    // 4. chunked delta scan (sequential depth 32 instead of 2048)
    delta_chunk_kernel<<<B * HH, 256, 0, stream>>>(qb, kb, vb, gate, am, yd, T);
    // 5. local attention
    local_attn_kernel<<<B * HH * (T / WIN), 64, 0, stream>>>(qb, kb, vb, am, yl, T);
    // 6. fuse + RMSNorm -> bf16
    combine_norm_kernel<<<BT, 256, 0, stream>>>(yd, yl, fuse, nw, ybf);
    // 7. output projection with fused residual epilogue
    {
        dim3 grid(BT / 64, EE / 256);
        wmma_gemm_bf16<<<grid, 256, 0, stream>>>(ybf, WoT, out, x, bo, gam, BT, EE, EE, 1);
    }
}
